// ClassConditionalCurvature_73005854097884
// MI455X (gfx1250) — matrix operand order repaired and checked by reading, not compiled, not run on hardware
//
#include <hip/hip_runtime.h>

// ============================================================================
// ClassConditionalCurvature loss for MI455X (gfx1250, wave32, WMMA).
//
// Roofline: dominant cost = Gram matrix E*E^T (N=8192, D=256) = 34.4 GFLOP
// + 67M-candidate masked top-k scan. 8MB embedding set is L2-resident
// (192MB L2) -> matrix-pipe bound, not HBM bound (23.3 TB/s).
// Strategy: bf16x3 split-precision WMMA (hi*hi + hi*lo + lo*hi, fp32 acc)
// for ~fp32-accurate dot products at bf16 WMMA rates; fused per-row top-k
// so the 268MB NxN matrix is never materialized. A fragments are hoisted
// into VGPRs (one-time load per WG); B tiles are staged through LDS once
// per WG per column step (4x cut of redundant per-wave L2 traffic).
// ============================================================================

#define NPTS 8192
#define DIM  256
#define NCLS 10
#define KNN  10
#define SPLITS 8
#define TM   128                 // rows per workgroup in fused GEMM
#define CPS  (NPTS / SPLITS)     // columns per split = 1024
#define EPSF 1e-8f
#define FINF 3.0e38f
#define BSTRIDE 264              // LDS B row stride (bf16): 132 dwords, %64==4

typedef __attribute__((ext_vector_type(16))) __bf16 bf16x16;
typedef __attribute__((ext_vector_type(8)))  __bf16 bf16x8;
typedef __attribute__((ext_vector_type(4)))  __bf16 bf16x4;
typedef __attribute__((ext_vector_type(8)))  float  v8f;

// ---- workspace layout (bytes) ----------------------------------------------
#define OFF_EHI   ((size_t)0)                               // bf16 hi  [N][D]
#define OFF_ELO   (OFF_EHI + (size_t)NPTS * DIM * 2)        // bf16 lo  [N][D]
#define OFF_SQN   (OFF_ELO + (size_t)NPTS * DIM * 2)        // f32 |e|^2 [N]
#define OFF_CENT  (OFF_SQN + (size_t)NPTS * 4)              // f32 [C][D] (sum->centroid)
#define OFF_CNT   (OFF_CENT + (size_t)NCLS * DIM * 4)       // f32 [16]
#define OFF_SPR   (OFF_CNT + 64)                            // f32 [16]
#define OFF_LOC   (OFF_SPR + 64)                            // f32 [128] (C*K sums)
#define OFF_KNNP  (OFF_LOC + 512)                           // f32 [SPLITS][N][KNN]
// total ~ 11.06 MB

// ---- zero the accumulator region -------------------------------------------
__global__ void zero_kernel(float* __restrict__ p) {
    const int n = (int)((OFF_KNNP - OFF_CENT) / 4);
    for (int i = threadIdx.x; i < n; i += 256) p[i] = 0.0f;
}

// ---- fp32 -> bf16 hi/lo split (planar) --------------------------------------
__global__ __launch_bounds__(256) void cvt_kernel(const float* __restrict__ emb,
                                                  __bf16* __restrict__ Ehi,
                                                  __bf16* __restrict__ Elo) {
    const int idx = blockIdx.x * 256 + threadIdx.x;      // over N*D/4
    float4 v = ((const float4*)emb)[idx];
    __bf16 h0 = (__bf16)v.x, h1 = (__bf16)v.y, h2 = (__bf16)v.z, h3 = (__bf16)v.w;
    bf16x4 hv = {h0, h1, h2, h3};
    bf16x4 lv = {(__bf16)(v.x - (float)h0), (__bf16)(v.y - (float)h1),
                 (__bf16)(v.z - (float)h2), (__bf16)(v.w - (float)h3)};
    ((bf16x4*)Ehi)[idx] = hv;
    ((bf16x4*)Elo)[idx] = lv;
}

// ---- squared norms ----------------------------------------------------------
__global__ __launch_bounds__(256) void sqn_kernel(const float* __restrict__ emb,
                                                  float* __restrict__ sqn) {
    const int i = blockIdx.x * 256 + threadIdx.x;
    const float4* e = (const float4*)(emb + (size_t)i * DIM);
    float acc = 0.0f;
#pragma unroll 8
    for (int d = 0; d < DIM / 4; ++d) {
        float4 v = e[d];
        acc += v.x * v.x + v.y * v.y + v.z * v.z + v.w * v.w;
    }
    sqn[i] = acc;
}

// ---- centroid sums + counts (thread == dim, LDS privatized) -----------------
__global__ __launch_bounds__(256) void csum_kernel(const float* __restrict__ emb,
                                                   const int* __restrict__ labels,
                                                   float* __restrict__ csum,
                                                   float* __restrict__ counts) {
    __shared__ float ls[NCLS * DIM];
    __shared__ float lc[16];
    const int tid = threadIdx.x;                          // == dim
    for (int i = tid; i < NCLS * DIM; i += 256) ls[i] = 0.0f;
    if (tid < 16) lc[tid] = 0.0f;
    __syncthreads();
    const int base = blockIdx.x * 128;                    // 128 points per WG
    for (int p = 0; p < 128; ++p) {
        const int lab = labels[base + p];                 // uniform -> scalar load
        ls[lab * DIM + tid] += emb[(size_t)(base + p) * DIM + tid];
    }
    if (tid < 128) atomicAdd(&lc[labels[base + tid]], 1.0f);
    __syncthreads();
    for (int i = tid; i < NCLS * DIM; i += 256) atomicAdd(&csum[i], ls[i]);
    if (tid < NCLS) atomicAdd(&counts[tid], lc[tid]);
}

// ---- centroid = sum / count -------------------------------------------------
__global__ void cdiv_kernel(float* __restrict__ cent, const float* __restrict__ counts) {
    const int idx = blockIdx.x * 256 + threadIdx.x;
    if (idx < NCLS * DIM) cent[idx] = cent[idx] / counts[idx >> 8];
}

// ---- per-class spread sums --------------------------------------------------
__global__ __launch_bounds__(256) void spread_kernel(const float* __restrict__ emb,
                                                     const int* __restrict__ labels,
                                                     const float* __restrict__ cent,
                                                     float* __restrict__ spr) {
    __shared__ float cls[16];
    const int tid = threadIdx.x;
    const int i = blockIdx.x * 256 + tid;
    if (tid < 16) cls[tid] = 0.0f;
    __syncthreads();
    const int lab = labels[i];
    const float4* e = (const float4*)(emb + (size_t)i * DIM);
    const float4* c = (const float4*)(cent + (size_t)lab * DIM);
    float acc = 0.0f;
#pragma unroll 8
    for (int d = 0; d < DIM / 4; ++d) {
        float4 a = e[d], b = c[d];
        float dx = a.x - b.x, dy = a.y - b.y, dz = a.z - b.z, dw = a.w - b.w;
        acc += dx * dx + dy * dy + dz * dz + dw * dw;
    }
    atomicAdd(&cls[lab], sqrtf(acc));
    __syncthreads();
    if (tid < NCLS) atomicAdd(&spr[tid], cls[tid]);
}

// ---- fused bf16x3 WMMA Gram + masked per-row top-10 -------------------------
// grid = (N/TM, SPLITS), block = 128 (4 wave32). Wave w computes two 16x16
// subtiles (rows rb+32w .. rb+32w+31) against a 16-column tile, K=256 in 8
// steps of 32, three v_wmma_f32_16x16x32_bf16 per step (hi*hi+hi*lo+lo*hi).
// A fragments: loaded once into VGPRs. B tile: staged via LDS once per WG.
__global__ __launch_bounds__(128) void gemm_topk_kernel(
    const __bf16* __restrict__ Ehi, const __bf16* __restrict__ Elo,
    const float* __restrict__ sqn, const int* __restrict__ labels,
    float* __restrict__ knnp)
{
    __shared__ float lds_dot[TM * 17];                       // padded, conflict-free
    __shared__ __attribute__((aligned(16))) __bf16 ldsBhi[16 * BSTRIDE];
    __shared__ __attribute__((aligned(16))) __bf16 ldsBlo[16 * BSTRIDE];
    __shared__ float lds_sqn[16];
    __shared__ int   lds_lab[16];

    const int tid   = threadIdx.x;
    const int lane  = tid & 31;
    const int w     = tid >> 5;
    const int rb    = blockIdx.x * TM;
    const int split = blockIdx.y;

    const int   r     = rb + tid;
    const int   lab_r = labels[r];
    const float sqn_r = sqn[r];

    float tk[KNN];
#pragma unroll
    for (int t = 0; t < KNN; ++t) tk[t] = FINF;

    // fragment lane decomposition (ISA 16-bit A 16x32 / B 32x16 layouts)
    const int fm = lane & 15;
    const int fh = lane >> 4;
    const size_t arow0 = (size_t)(rb + w * 32 + fm) * DIM;        // subtile 0 row
    const size_t arow1 = (size_t)(rb + w * 32 + 16 + fm) * DIM;   // subtile 1 row

    // ---- hoist all A fragments (ct-invariant) into VGPRs ----
    bf16x16 A0h[8], A0l[8], A1h[8], A1l[8];
#pragma unroll
    for (int kk = 0; kk < 8; ++kk) {
        const int ao = kk * 32 + 8 * fh;   // elems 0..7 -> K=k0+8fh+0..7; 8..15 -> +16
        bf16x8 h0, h1;
        h0 = *(const bf16x8*)(Ehi + arow0 + ao);
        h1 = *(const bf16x8*)(Ehi + arow0 + ao + 16);
        A0h[kk] = __builtin_shufflevector(h0, h1, 0,1,2,3,4,5,6,7,8,9,10,11,12,13,14,15);
        h0 = *(const bf16x8*)(Elo + arow0 + ao);
        h1 = *(const bf16x8*)(Elo + arow0 + ao + 16);
        A0l[kk] = __builtin_shufflevector(h0, h1, 0,1,2,3,4,5,6,7,8,9,10,11,12,13,14,15);
        h0 = *(const bf16x8*)(Ehi + arow1 + ao);
        h1 = *(const bf16x8*)(Ehi + arow1 + ao + 16);
        A1h[kk] = __builtin_shufflevector(h0, h1, 0,1,2,3,4,5,6,7,8,9,10,11,12,13,14,15);
        h0 = *(const bf16x8*)(Elo + arow1 + ao);
        h1 = *(const bf16x8*)(Elo + arow1 + ao + 16);
        A1l[kk] = __builtin_shufflevector(h0, h1, 0,1,2,3,4,5,6,7,8,9,10,11,12,13,14,15);
    }

    // B staging decomposition: 16 rows x 512B(hi)+512B(lo), 8 chunks of 64B/row
    const int sc  = tid >> 3;          // staging row 0..15
    const int sch = (tid & 7) * 32;    // element offset within row

    // LDS fragment read base (per documented B 32x16 layout: n=lane&15,
    // elems 0..7 -> K 8fh+0..7, elems 8..15 -> K 16+8fh+0..7)
    const __bf16* bh_base = ldsBhi + fm * BSTRIDE + 8 * fh;
    const __bf16* bl_base = ldsBlo + fm * BSTRIDE + 8 * fh;

    const v8f vzero = {0.f, 0.f, 0.f, 0.f, 0.f, 0.f, 0.f, 0.f};

    for (int ct = 0; ct < CPS / 16; ++ct) {
        const int cb = split * CPS + ct * 16;

        __syncthreads();               // prior scan done: B + dot regions free
        {   // stage B tile (hi+lo) into padded-row LDS
            const __bf16* gh = Ehi + (size_t)(cb + sc) * DIM + sch;
            const __bf16* gl = Elo + (size_t)(cb + sc) * DIM + sch;
            __bf16* shp = ldsBhi + sc * BSTRIDE + sch;
            __bf16* slp = ldsBlo + sc * BSTRIDE + sch;
#pragma unroll
            for (int q = 0; q < 4; ++q) {
                *(bf16x8*)(shp + q * 8) = *(const bf16x8*)(gh + q * 8);
                *(bf16x8*)(slp + q * 8) = *(const bf16x8*)(gl + q * 8);
            }
        }
        if (tid < 16) {
            lds_lab[tid] = labels[cb + tid];
            lds_sqn[tid] = sqn[cb + tid];
        }
        __syncthreads();               // B tile ready

        // 4 independent accumulation chains: (subtile, kk parity)
        v8f acc[4];
#pragma unroll
        for (int q = 0; q < 4; ++q) acc[q] = vzero;

#pragma unroll
        for (int kk = 0; kk < 8; ++kk) {
            const int k0 = kk * 32;
            bf16x8 b0 = *(const bf16x8*)(bh_base + k0);
            bf16x8 b1 = *(const bf16x8*)(bh_base + k0 + 16);
            bf16x8 c0 = *(const bf16x8*)(bl_base + k0);
            bf16x8 c1 = *(const bf16x8*)(bl_base + k0 + 16);
            bf16x16 Bh = __builtin_shufflevector(b0, b1, 0,1,2,3,4,5,6,7,8,9,10,11,12,13,14,15);
            bf16x16 Bl = __builtin_shufflevector(c0, c1, 0,1,2,3,4,5,6,7,8,9,10,11,12,13,14,15);
            const int q = kk & 1;
            acc[q]     = __builtin_amdgcn_wmma_f32_16x16x32_bf16(false, A0h[kk], false, Bh, (short)0, acc[q],     false, false);
            acc[2 + q] = __builtin_amdgcn_wmma_f32_16x16x32_bf16(false, A1h[kk], false, Bh, (short)0, acc[2 + q], false, false);
            acc[q]     = __builtin_amdgcn_wmma_f32_16x16x32_bf16(false, A0h[kk], false, Bl, (short)0, acc[q],     false, false);
            acc[2 + q] = __builtin_amdgcn_wmma_f32_16x16x32_bf16(false, A1h[kk], false, Bl, (short)0, acc[2 + q], false, false);
            acc[q]     = __builtin_amdgcn_wmma_f32_16x16x32_bf16(false, A0l[kk], false, Bh, (short)0, acc[q],     false, false);
            acc[2 + q] = __builtin_amdgcn_wmma_f32_16x16x32_bf16(false, A1l[kk], false, Bh, (short)0, acc[2 + q], false, false);
        }

        // write dot tile. C/D layout: vgpr v, lanes 0-15 -> (M=v, N=lane);
        // lanes 16-31 -> (M=v+8, N=lane-16)
#pragma unroll
        for (int v = 0; v < 8; ++v) {
            lds_dot[(w * 32 + v + 8 * fh) * 17 + fm]      = acc[0][v] + acc[1][v];
            lds_dot[(w * 32 + 16 + v + 8 * fh) * 17 + fm] = acc[2][v] + acc[3][v];
        }
        __syncthreads();               // dot tile ready

        // per-row masked top-10 (self excluded == reference's drop-smallest)
#pragma unroll 4
        for (int j = 0; j < 16; ++j) {
            if (lds_lab[j] == lab_r) {
                const int jj = cb + j;
                if (jj != r) {
                    float d2 = fmaxf(sqn_r + lds_sqn[j] - 2.0f * lds_dot[tid * 17 + j], 0.0f);
                    if (d2 < tk[KNN - 1]) {
                        tk[KNN - 1] = d2;
#pragma unroll
                        for (int s = KNN - 1; s > 0; --s) {
                            float a = tk[s - 1], b = tk[s];
                            tk[s - 1] = fminf(a, b);
                            tk[s]     = fmaxf(a, b);
                        }
                    }
                }
            }
        }
    }

    float* outp = knnp + ((size_t)split * NPTS + r) * KNN;
#pragma unroll
    for (int t = 0; t < KNN; ++t) outp[t] = tk[t];
}

// ---- merge 8 sorted partial lists, normalize, per-class local sums ----------
__global__ __launch_bounds__(128) void merge_kernel(const float* __restrict__ knnp,
                                                    const int* __restrict__ labels,
                                                    float* __restrict__ locsum) {
    __shared__ float ls[NCLS * KNN];
    const int tid = threadIdx.x;
    if (tid < NCLS * KNN) ls[tid] = 0.0f;
    __syncthreads();

    const int i = blockIdx.x * 128 + tid;
    int p[SPLITS];
#pragma unroll
    for (int l = 0; l < SPLITS; ++l) p[l] = 0;

    float knn[KNN];
    float sum = 0.0f;
#pragma unroll
    for (int sel = 0; sel < KNN; ++sel) {
        float best = FINF;
        int bl = 0;
#pragma unroll
        for (int l = 0; l < SPLITS; ++l) {
            float h = (p[l] < KNN) ? knnp[((size_t)l * NPTS + i) * KNN + p[l]] : FINF;
            if (h < best) { best = h; bl = l; }
        }
#pragma unroll
        for (int l = 0; l < SPLITS; ++l) if (bl == l) p[l]++;
        float dk = sqrtf(fmaxf(best, 0.0f));
        knn[sel] = dk;
        sum += dk;
    }
    const float inv = 1.0f / (sum * (1.0f / KNN) + EPSF);
    const int lab = labels[i];
#pragma unroll
    for (int t = 0; t < KNN; ++t) atomicAdd(&ls[lab * KNN + t], knn[t] * inv);
    __syncthreads();
    if (tid < NCLS * KNN) atomicAdd(&locsum[tid], ls[tid]);
}

// ---- final: inter-class structure + all MSE terms ---------------------------
__global__ __launch_bounds__(128) void final_kernel(
    const float* __restrict__ cent, const float* __restrict__ counts,
    const float* __restrict__ spr,  const float* __restrict__ locsum,
    const float* __restrict__ ref_d, const float* __restrict__ ref_a,
    const float* __restrict__ ref_s, const float* __restrict__ ref_l,
    float* __restrict__ out)
{
    __shared__ float mu[DIM];
    __shared__ float nrm[NCLS];
    __shared__ float dst[100];
    __shared__ float ang[100];
    const int tid = threadIdx.x;

    for (int d = tid; d < DIM; d += 128) {
        float s = 0.0f;
        for (int c = 0; c < NCLS; ++c) s += cent[c * DIM + d];
        mu[d] = s * (1.0f / NCLS);
    }
    __syncthreads();
    if (tid < NCLS) {
        float s = 0.0f;
        for (int d = 0; d < DIM; ++d) { float x = cent[tid * DIM + d] - mu[d]; s += x * x; }
        nrm[tid] = fmaxf(sqrtf(s), EPSF);
    }
    __syncthreads();
    if (tid < 100) {
        const int a = tid / 10, b = tid % 10;
        float s = 0.0f, t = 0.0f;
        for (int d = 0; d < DIM; ++d) {
            float xa = cent[a * DIM + d], xb = cent[b * DIM + d];
            float df = xa - xb;
            s += df * df;
            t += (xa - mu[d]) * (xb - mu[d]);
        }
        dst[tid] = (s > 0.0f) ? sqrtf(s) : 0.0f;   // safe_sqrt semantics
        ang[tid] = t / (nrm[a] * nrm[b]);
    }
    __syncthreads();
    if (tid == 0) {
        float dm = 0.0f, rdm = 0.0f;
        for (int t = 0; t < 100; ++t) { dm += dst[t]; rdm += ref_d[t]; }
        dm *= 0.01f; rdm *= 0.01f;
        const float idm = 1.0f / (dm + EPSF), irdm = 1.0f / (rdm + EPSF);
        float cl = 0.0f, al = 0.0f;
        for (int t = 0; t < 100; ++t) {
            float df = dst[t] * idm - ref_d[t] * irdm; cl += df * df;
            float da = ang[t] - ref_a[t];              al += da * da;
        }
        cl *= 0.01f; al *= 0.01f;

        float sp[NCLS];
        float sm = 0.0f, rsm = 0.0f;
        for (int c = 0; c < NCLS; ++c) { sp[c] = spr[c] / counts[c]; sm += sp[c]; rsm += ref_s[c]; }
        sm *= 0.1f; rsm *= 0.1f;
        const float ism = 1.0f / (sm + EPSF), irsm = 1.0f / (rsm + EPSF);
        float sl = 0.0f;
        for (int c = 0; c < NCLS; ++c) { float df = sp[c] * ism - ref_s[c] * irsm; sl += df * df; }
        sl *= 0.1f;

        float ll = 0.0f;
        for (int t = 0; t < 100; ++t) { float df = locsum[t] / counts[t / 10] - ref_l[t]; ll += df * df; }
        ll *= 0.01f;

        out[0] = 1.0f * (cl + al) + 0.5f * sl + 0.5f * ll;
    }
}

// ============================================================================
extern "C" void kernel_launch(void* const* d_in, const int* in_sizes, int n_in,
                              void* d_out, int out_size, void* d_ws, size_t ws_size,
                              hipStream_t stream) {
    const float* emb    = (const float*)d_in[0];
    const int*   labels = (const int*)d_in[1];
    const float* ref_d  = (const float*)d_in[2];
    const float* ref_a  = (const float*)d_in[3];
    const float* ref_s  = (const float*)d_in[4];
    const float* ref_l  = (const float*)d_in[5];

    char* ws = (char*)d_ws;
    __bf16* Ehi = (__bf16*)(ws + OFF_EHI);
    __bf16* Elo = (__bf16*)(ws + OFF_ELO);
    float*  sqn = (float*)(ws + OFF_SQN);
    float*  cen = (float*)(ws + OFF_CENT);
    float*  cnt = (float*)(ws + OFF_CNT);
    float*  spr = (float*)(ws + OFF_SPR);
    float*  loc = (float*)(ws + OFF_LOC);
    float*  knp = (float*)(ws + OFF_KNNP);

    hipLaunchKernelGGL(zero_kernel,   dim3(1),                     dim3(256), 0, stream, cen);
    hipLaunchKernelGGL(cvt_kernel,    dim3(NPTS * DIM / 4 / 256),  dim3(256), 0, stream, emb, Ehi, Elo);
    hipLaunchKernelGGL(sqn_kernel,    dim3(NPTS / 256),            dim3(256), 0, stream, emb, sqn);
    hipLaunchKernelGGL(csum_kernel,   dim3(NPTS / 128),            dim3(256), 0, stream, emb, labels, cen, cnt);
    hipLaunchKernelGGL(cdiv_kernel,   dim3(10),                    dim3(256), 0, stream, cen, cnt);
    hipLaunchKernelGGL(spread_kernel, dim3(NPTS / 256),            dim3(256), 0, stream, emb, labels, cen, spr);
    hipLaunchKernelGGL(gemm_topk_kernel, dim3(NPTS / TM, SPLITS),  dim3(128), 0, stream, Ehi, Elo, sqn, labels, knp);
    hipLaunchKernelGGL(merge_kernel,  dim3(NPTS / 128),            dim3(128), 0, stream, knp, labels, loc);
    hipLaunchKernelGGL(final_kernel,  dim3(1),                     dim3(128), 0, stream,
                       cen, cnt, spr, loc, ref_d, ref_a, ref_s, ref_l, (float*)d_out);

    (void)in_sizes; (void)n_in; (void)out_size; (void)ws_size;
}
